// EdrakAttention_39917426049504
// MI455X (gfx1250) — matrix operand order repaired and checked by reading, not compiled, not run on hardware
//
#include <hip/hip_runtime.h>
#include <hip/hip_bf16.h>

// ---------------------------------------------------------------------------
// MI455X (gfx1250) attention layer: bf16 WMMA GEMMs (TDM-staged B) + fused RoPE
// + flash attention.
// ---------------------------------------------------------------------------

typedef __attribute__((ext_vector_type(16))) __bf16 v16bf;
typedef __attribute__((ext_vector_type(8)))  __bf16 v8bf;
typedef __attribute__((ext_vector_type(8)))  float  v8f;
typedef __attribute__((ext_vector_type(4)))  unsigned v4u;
typedef __attribute__((ext_vector_type(8)))  int      v8i;
typedef __attribute__((ext_vector_type(4)))  int      v4i;

#define WMMA_BF16(a, b, c) \
  __builtin_amdgcn_wmma_f32_16x16x32_bf16(false, (a), false, (b), (short)0, (c), false, false)

__device__ __forceinline__ __bf16 f2bf(float x) {
  unsigned u = __builtin_bit_cast(unsigned, x);
  u += 0x7fffu + ((u >> 16) & 1u);            // round-to-nearest-even
  unsigned short h = (unsigned short)(u >> 16);
  return __builtin_bit_cast(__bf16, h);
}

__device__ __forceinline__ v16bf cat8(v8bf a, v8bf b) {
  return __builtin_shufflevector(a, b, 0,1,2,3,4,5,6,7,8,9,10,11,12,13,14,15);
}

static constexpr int S_LEN   = 2048;
static constexpr int K_DIM   = 2048;   // inner dim of every GEMM here
static constexpr int HEAD_D  = 64;
static constexpr int KCHUNK  = 128;    // K halves staged per TDM transfer (16 KB)

enum { MODE_Q = 0, MODE_K = 1, MODE_V = 2, MODE_O = 3 };

// ---------------------------------------------------------------------------
// fp32 -> bf16 conversion (grid-stride)
// ---------------------------------------------------------------------------
__global__ void cvt_f32_bf16(const float* __restrict__ src, __bf16* __restrict__ dst, int n) {
  int i = blockIdx.x * blockDim.x + threadIdx.x;
  int stride = gridDim.x * blockDim.x;
  for (; i < n; i += stride) dst[i] = f2bf(src[i]);
}

// ---------------------------------------------------------------------------
// TDM: stage a 64-row x KCHUNK-half strip of W (row-major [N][K] bf16) into LDS.
// Issued by one wave; completion tracked with TENSORcnt.
// ---------------------------------------------------------------------------
__device__ __forceinline__ void tdm_load_b_strip(const __bf16* gsrc, unsigned ldsAddr,
                                                 int nRows, int kLen) {
#if __has_builtin(__builtin_amdgcn_tensor_load_to_lds)
  unsigned long long ga = (unsigned long long)(const void*)gsrc;
  v4u g0;
  g0.x = 1u;                                   // count=1, is_restore=0, gather off
  g0.y = ldsAddr;                              // LDS byte address
  g0.z = (unsigned)ga;                         // global_addr[31:0]
  g0.w = (unsigned)(ga >> 32) | (2u << 30);    // global_addr[56:32] | type=2 (image)

  v8i g1;
  g1[0] = (int)(1u << 16);                               // data_size = 1 -> 2 bytes
  g1[1] = (int)(((unsigned)K_DIM & 0xffffu) << 16);      // tensor_dim0[15:0]
  g1[2] = (int)((((unsigned)K_DIM >> 16) & 0xffffu) |    // tensor_dim0[31:16]
                (((unsigned)nRows & 0xffffu) << 16));    // tensor_dim1[15:0]
  g1[3] = (int)((((unsigned)nRows >> 16) & 0xffffu) |    // tensor_dim1[31:16]
                (((unsigned)kLen & 0xffffu) << 16));     // tile_dim0
  g1[4] = (int)(64u);                                    // tile_dim1 = 64 rows
  g1[5] = (int)((unsigned)K_DIM);                        // tensor_dim0_stride[31:0]
  g1[6] = 0;                                             // stride hi / dim1_stride lo
  g1[7] = 0;

  v4i gz = {0, 0, 0, 0};
#if __clang_major__ >= 23
  v8i gz8 = {0, 0, 0, 0, 0, 0, 0, 0};
  __builtin_amdgcn_tensor_load_to_lds(g0, g1, gz, gz, gz8, 0);
#else
  __builtin_amdgcn_tensor_load_to_lds(g0, g1, gz, gz, 0);
#endif
  __builtin_amdgcn_s_wait_tensorcnt(0);
#else
  (void)gsrc; (void)ldsAddr; (void)nRows; (void)kLen;
#endif
}

// ---------------------------------------------------------------------------
// Generic bf16 WMMA GEMM: out[m,n] = sum_k A[m,k] * W[n,k]  (W is [N,K] row-major)
// wave tile: 16 (M) x 64 (N); block = 8 waves -> 128 rows x 64 cols.
// The 64x K_DIM B strip is shared by all 8 waves -> staged through LDS by the
// Tensor Data Mover in KCHUNK slices; waves read B fragments with ds_load.
// Epilogues: MODE_Q/K fused RoPE + [head][s][d] bf16; MODE_V transposed bf16;
//            MODE_O fp32 [s][n] to d_out.
// ---------------------------------------------------------------------------
__global__ __launch_bounds__(256)
void gemm_qkvo(const __bf16* __restrict__ A, const __bf16* __restrict__ W,
               __bf16* __restrict__ outB, float* __restrict__ outF, int N, int mode) {
  __shared__ __align__(128) __bf16 ldsB[64 * KCHUNK];   // 16 KB B strip slice

  const int lane   = threadIdx.x & 31;
  const int wave   = threadIdx.x >> 5;
  const int lo16   = lane & 15;
  const bool hiH   = lane >= 16;
  const int rowBase = blockIdx.x * 128 + wave * 16;
  const int colBase = blockIdx.y * 64;

  v8f acc0 = {}, acc1 = {}, acc2 = {}, acc3 = {};

  // A fragment addressing (16-bit A 16x32 layout): lane row = lane&15,
  // halves 0..15 -> K = kSel+0..15 (kSel = hi? 16:0)
  const int kSelA = hiH ? 16 : 0;
  const __bf16* aPtr = A + (size_t)(rowBase + lo16) * K_DIM + kSelA;

  // B fragment addressing (16-bit B 32x16): lane col = lane&15,
  // halves i -> K = kOff + i (kOff = hi? 16:0): one contiguous 32B LDS read per tile.
  const int kOffB = hiH ? 16 : 0;
  const __bf16* lb0 = &ldsB[(size_t)( 0 + lo16) * KCHUNK + kOffB];
  const __bf16* lb1 = &ldsB[(size_t)(16 + lo16) * KCHUNK + kOffB];
  const __bf16* lb2 = &ldsB[(size_t)(32 + lo16) * KCHUNK + kOffB];
  const __bf16* lb3 = &ldsB[(size_t)(48 + lo16) * KCHUNK + kOffB];

  const unsigned ldsAddr = (unsigned)(unsigned long long)(const void*)&ldsB[0];

  for (int kc = 0; kc < K_DIM; kc += KCHUNK) {
    __syncthreads();   // previous slice fully consumed
#if __has_builtin(__builtin_amdgcn_tensor_load_to_lds)
    if (wave == 0) {
      tdm_load_b_strip(W + (size_t)colBase * K_DIM + kc, ldsAddr, N, KCHUNK);
    }
#else
    // fallback: cooperative copy, 256 threads x v8bf
    for (int i = threadIdx.x; i < 64 * KCHUNK / 8; i += 256) {
      int h = i * 8, rw = h / KCHUNK, kk = h % KCHUNK;
      *(v8bf*)&ldsB[rw * KCHUNK + kk] =
          *(const v8bf*)(W + (size_t)(colBase + rw) * K_DIM + kc + kk);
    }
#endif
    __syncthreads();   // slice visible to all waves

#pragma unroll
    for (int k0 = 0; k0 < KCHUNK; k0 += 32) {
      v8bf alo = *(const v8bf*)(aPtr + kc + k0);
      v8bf ahi = *(const v8bf*)(aPtr + kc + k0 + 8);
      v16bf a  = cat8(alo, ahi);
      v16bf b0 = *(const v16bf*)(lb0 + k0);
      v16bf b1 = *(const v16bf*)(lb1 + k0);
      v16bf b2 = *(const v16bf*)(lb2 + k0);
      v16bf b3 = *(const v16bf*)(lb3 + k0);
      acc0 = WMMA_BF16(a, b0, acc0);
      acc1 = WMMA_BF16(a, b1, acc1);
      acc2 = WMMA_BF16(a, b2, acc2);
      acc3 = WMMA_BF16(a, b3, acc3);
    }
  }

  const int mB = hiH ? 8 : 0;

  if (mode == MODE_Q || mode == MODE_K) {
    // RoPE: within this wave's 64 columns (= one head), d = 16*t + lo16,
    // rotate-half partner d+32 lives in accumulator t+2, same lane & register.
    const float LN_BASE_OVER_32 = 0.28782313662425572f;  // ln(10000)/32
    const float if0 = __expf(-(float)(lo16)      * LN_BASE_OVER_32);
    const float if1 = __expf(-(float)(16 + lo16) * LN_BASE_OVER_32);
#pragma unroll
    for (int r = 0; r < 8; ++r) {
      const float s = (float)(rowBase + mB + r);
      float c, si;
      __sincosf(s * if0, &si, &c);
      { float x = acc0[r], y = acc2[r]; acc0[r] = x * c - y * si; acc2[r] = y * c + x * si; }
      __sincosf(s * if1, &si, &c);
      { float x = acc1[r], y = acc3[r]; acc1[r] = x * c - y * si; acc3[r] = y * c + x * si; }
    }
  }

#pragma unroll
  for (int r = 0; r < 8; ++r) {
    const int m  = rowBase + mB + r;
    const int c0 = colBase + lo16;
    float v0 = acc0[r], v1 = acc1[r], v2 = acc2[r], v3 = acc3[r];
    if (mode == MODE_O) {
      outF[(size_t)m * 2048 + c0]      = v0;
      outF[(size_t)m * 2048 + c0 + 16] = v1;
      outF[(size_t)m * 2048 + c0 + 32] = v2;
      outF[(size_t)m * 2048 + c0 + 48] = v3;
    } else if (mode == MODE_V) {
      // transposed: [col][s]  (col = kvh*64 + d)
      outB[(size_t)(c0)      * S_LEN + m] = f2bf(v0);
      outB[(size_t)(c0 + 16) * S_LEN + m] = f2bf(v1);
      outB[(size_t)(c0 + 32) * S_LEN + m] = f2bf(v2);
      outB[(size_t)(c0 + 48) * S_LEN + m] = f2bf(v3);
    } else {
      // [head][s][d]
      int cc[4] = { c0, c0 + 16, c0 + 32, c0 + 48 };
      float vv[4] = { v0, v1, v2, v3 };
#pragma unroll
      for (int t = 0; t < 4; ++t) {
        int h = cc[t] >> 6, d = cc[t] & 63;
        outB[((size_t)h * S_LEN + m) * HEAD_D + d] = f2bf(vv[t]);
      }
    }
  }
}

// ---------------------------------------------------------------------------
// Flash attention forward: wave = 16 query rows of one head; key blocks of 32.
// Q: [head][s][64] bf16 (RoPE'd), K: [kvh][s][64] bf16 (RoPE'd), VT: [kvh][64][s] bf16.
// ctx out: [s][head*64+d] bf16.
// ---------------------------------------------------------------------------
__global__ __launch_bounds__(256)
void attn_fwd(const __bf16* __restrict__ Q, const __bf16* __restrict__ Kc,
              const __bf16* __restrict__ VT, __bf16* __restrict__ ctx) {
  __shared__ __align__(16) __bf16 ldsP[8][16][48];  // per-wave P staging (padded)

  const int lane = threadIdx.x & 31;
  const int wave = threadIdx.x >> 5;
  const int lo16 = lane & 15;
  const bool hiH = lane >= 16;
  const int kSelA = hiH ? 16 : 0;
  const int kOffB = hiH ? 16 : 0;
  const int mB    = hiH ? 8 : 0;

  const int head  = blockIdx.y;
  const int kvh   = head >> 2;
  const int qRow0 = blockIdx.x * 128 + wave * 16;

  // Q A-fragments (K-dim 64 => two 16x32 fragments)
  const __bf16* qp = Q + ((size_t)head * S_LEN + qRow0 + lo16) * HEAD_D + kSelA;
  v16bf aq0 = cat8(*(const v8bf*)(qp + 0),  *(const v8bf*)(qp + 8));
  v16bf aq1 = cat8(*(const v8bf*)(qp + 32), *(const v8bf*)(qp + 40));

  float mrow[8], lrow[8];
#pragma unroll
  for (int r = 0; r < 8; ++r) { mrow[r] = -3.0e38f; lrow[r] = 0.f; }
  v8f o0 = {}, o1 = {}, o2 = {}, o3 = {};

  const __bf16* kbase = Kc + (size_t)kvh * S_LEN * HEAD_D;
  const __bf16* vbase = VT + (size_t)kvh * HEAD_D * S_LEN;

  const int kEnd = qRow0 + 15;  // causal limit for this wave
  for (int keyBase = 0; keyBase <= kEnd; keyBase += 32) {
    if (keyBase + 32 <= kEnd) {  // prefetch next K/V block into cache
      __builtin_prefetch(kbase + (size_t)(keyBase + 32) * HEAD_D, 0, 3);
      __builtin_prefetch(vbase + (keyBase + 32), 0, 3);
    }

    // ---- scores = Q * K^T (two 16-key tiles, two K-dim steps each) ----
    v8f s0 = {}, s1 = {};
    {
      const __bf16* kp0 = kbase + (size_t)(keyBase + lo16) * HEAD_D;
      const __bf16* kp1 = kbase + (size_t)(keyBase + 16 + lo16) * HEAD_D;
      v16bf b00 = *(const v16bf*)(kp0 + kOffB);
      v16bf b01 = *(const v16bf*)(kp0 + 32 + kOffB);
      v16bf b10 = *(const v16bf*)(kp1 + kOffB);
      v16bf b11 = *(const v16bf*)(kp1 + 32 + kOffB);
      s0 = WMMA_BF16(aq0, b00, s0);
      s0 = WMMA_BF16(aq1, b01, s0);
      s1 = WMMA_BF16(aq0, b10, s1);
      s1 = WMMA_BF16(aq1, b11, s1);
    }

    // ---- online softmax (row reductions via shfl_xor within 16-lane halves) ----
#pragma unroll
    for (int r = 0; r < 8; ++r) {
      const int m = qRow0 + mB + r;
      const int key0 = keyBase + lo16;
      float x0 = s0[r] * 0.125f; if (key0 > m)      x0 = -3.0e38f;
      float x1 = s1[r] * 0.125f; if (key0 + 16 > m) x1 = -3.0e38f;
      float mx = fmaxf(x0, x1);
      mx = fmaxf(mx, __shfl_xor(mx, 1, 32));
      mx = fmaxf(mx, __shfl_xor(mx, 2, 32));
      mx = fmaxf(mx, __shfl_xor(mx, 4, 32));
      mx = fmaxf(mx, __shfl_xor(mx, 8, 32));
      const float mnew = fmaxf(mrow[r], mx);
      const float p0 = __expf(x0 - mnew);
      const float p1 = __expf(x1 - mnew);
      float ps = p0 + p1;
      ps += __shfl_xor(ps, 1, 32);
      ps += __shfl_xor(ps, 2, 32);
      ps += __shfl_xor(ps, 4, 32);
      ps += __shfl_xor(ps, 8, 32);
      const float alpha = __expf(mrow[r] - mnew);
      lrow[r] = lrow[r] * alpha + ps;
      mrow[r] = mnew;
      o0[r] *= alpha; o1[r] *= alpha; o2[r] *= alpha; o3[r] *= alpha;
      // stage P (C-layout) into LDS for A-layout reload
      ldsP[wave][mB + r][lo16]      = f2bf(p0);
      ldsP[wave][mB + r][16 + lo16] = f2bf(p1);
    }
    asm volatile("s_wait_dscnt 0" ::: "memory");

    // reload P as a 16x32 A-fragment
    v8bf plo = *(const v8bf*)(&ldsP[wave][lo16][kSelA]);
    v8bf phi = *(const v8bf*)(&ldsP[wave][lo16][kSelA + 8]);
    v16bf pA = cat8(plo, phi);

    // ---- O += P * V (V pre-transposed -> contiguous B fragments) ----
    const __bf16* vp = vbase + keyBase + kOffB;
    v16bf bv0 = *(const v16bf*)(vp + (size_t)(0  + lo16) * S_LEN);
    v16bf bv1 = *(const v16bf*)(vp + (size_t)(16 + lo16) * S_LEN);
    v16bf bv2 = *(const v16bf*)(vp + (size_t)(32 + lo16) * S_LEN);
    v16bf bv3 = *(const v16bf*)(vp + (size_t)(48 + lo16) * S_LEN);
    o0 = WMMA_BF16(pA, bv0, o0);
    o1 = WMMA_BF16(pA, bv1, o1);
    o2 = WMMA_BF16(pA, bv2, o2);
    o3 = WMMA_BF16(pA, bv3, o3);
  }

  // normalize and write context [s][head*64 + d]
#pragma unroll
  for (int r = 0; r < 8; ++r) {
    const int m = qRow0 + mB + r;
    const float inv = 1.0f / lrow[r];
    __bf16* cp = ctx + (size_t)m * 2048 + head * HEAD_D + lo16;
    cp[0]  = f2bf(o0[r] * inv);
    cp[16] = f2bf(o1[r] * inv);
    cp[32] = f2bf(o2[r] * inv);
    cp[48] = f2bf(o3[r] * inv);
  }
}

// ---------------------------------------------------------------------------
// Launch
// ---------------------------------------------------------------------------
extern "C" void kernel_launch(void* const* d_in, const int* in_sizes, int n_in,
                              void* d_out, int out_size, void* d_ws, size_t ws_size,
                              hipStream_t stream) {
  const float* hX  = (const float*)d_in[0];
  const float* hWq = (const float*)d_in[1];
  const float* hWk = (const float*)d_in[2];
  const float* hWv = (const float*)d_in[3];
  const float* hWo = (const float*)d_in[4];

  char* ws = (char*)d_ws;
  const size_t MB = 1024 * 1024;
  __bf16* Xb  = (__bf16*)(ws + 0);          //  8 MiB : [2048][2048]
  __bf16* Wqb = (__bf16*)(ws + 8  * MB);    //  8 MiB : [2048][2048]
  __bf16* Wkb = (__bf16*)(ws + 16 * MB);    //  2 MiB : [512][2048]
  __bf16* Wvb = (__bf16*)(ws + 18 * MB);    //  2 MiB : [512][2048]
  __bf16* Wob = (__bf16*)(ws + 20 * MB);    //  8 MiB : [2048][2048]
  __bf16* Qb  = (__bf16*)(ws + 28 * MB);    //  8 MiB : [32][2048][64]
  __bf16* Kb  = (__bf16*)(ws + 36 * MB);    //  2 MiB : [8][2048][64]
  __bf16* VTb = (__bf16*)(ws + 38 * MB);    //  2 MiB : [8][64][2048]
  __bf16* Ctx = (__bf16*)(ws + 40 * MB);    //  8 MiB : [2048][2048]

  dim3 cb(256);
  cvt_f32_bf16<<<1024, cb, 0, stream>>>(hX,  Xb,  2048 * 2048);
  cvt_f32_bf16<<<1024, cb, 0, stream>>>(hWq, Wqb, 2048 * 2048);
  cvt_f32_bf16<<<512,  cb, 0, stream>>>(hWk, Wkb, 512 * 2048);
  cvt_f32_bf16<<<512,  cb, 0, stream>>>(hWv, Wvb, 512 * 2048);
  cvt_f32_bf16<<<1024, cb, 0, stream>>>(hWo, Wob, 2048 * 2048);

  dim3 blk(256);
  gemm_qkvo<<<dim3(16, 32), blk, 0, stream>>>(Xb, Wqb, Qb,  nullptr, 2048, MODE_Q);
  gemm_qkvo<<<dim3(16, 8),  blk, 0, stream>>>(Xb, Wkb, Kb,  nullptr, 512,  MODE_K);
  gemm_qkvo<<<dim3(16, 8),  blk, 0, stream>>>(Xb, Wvb, VTb, nullptr, 512,  MODE_V);

  attn_fwd<<<dim3(16, 32), blk, 0, stream>>>(Qb, Kb, VTb, Ctx);

  gemm_qkvo<<<dim3(16, 32), blk, 0, stream>>>(Ctx, Wob, nullptr, (float*)d_out, 2048, MODE_O);
}
